// PUTEncoder_78288663872050
// MI455X (gfx1250) — matrix operand order repaired and checked by reading, not compiled, not run on hardware
//
#include <hip/hip_runtime.h>
#include <hip/hip_bf16.h>

typedef _Float16 half_t;
typedef __attribute__((ext_vector_type(16))) _Float16 v16h;
typedef __attribute__((ext_vector_type(8)))  _Float16 v8h;
typedef __attribute__((ext_vector_type(8)))  float    v8f;

#define EPSV 1e-5f
constexpr int Bsz = 4, Npt = 2048, CinC = 64, CoutC = 256, KNB = 21, HEADS = 7;
constexpr int NPT  = Bsz * Npt;                 // 8192 points
constexpr long long ROWS = (long long)NPT * KNB; // 172032 gather rows

// ---------------- WMMA helpers (CDNA5 16x16x32 f16 -> f32) ----------------

__device__ __forceinline__ v8f wmma32(v16h a, v16h b, v8f c) {
  return __builtin_amdgcn_wmma_f32_16x16x32_f16(false, a, false, b, (short)0, c,
                                                false, false);
}

// A fragment: 16x32 f16 tile, row-major source with leading dim ld (halves).
// Contiguous K order used here: lane l holds row m=l&15; lanes 0-15 K=0-15; lanes 16-31 K=16-31.
__device__ __forceinline__ v16h load_a_frag(const half_t* A, int ld, int kk) {
  int lane = threadIdx.x & 31;
  int m = lane & 15;
  int kb = (lane >> 4) << 4;                       // 0 or 16
  const half_t* p = A + (size_t)m * ld + kk + kb;
  v8h lo = *(const v8h*)p;                         // K = kb .. kb+7
  v8h hi = *(const v8h*)(p + 8);                   // K = kb+8 .. kb+15
  v16h r;
#pragma unroll
  for (int i = 0; i < 8; ++i) { r[i] = lo[i]; r[i + 8] = hi[i]; }
  return r;
}

// B fragment where logical B[k][n] = W[nbase+n][kk+k], W row-major [O][ld].
// Contiguous K order used here: lane l holds col n=l&15; lanes 0-15 K=0-15, lanes 16-31 K=16-31.
__device__ __forceinline__ v16h load_b_rowT(const half_t* W, int ld, int kk, int nbase) {
  int lane = threadIdx.x & 31;
  const half_t* p = W + (size_t)(nbase + (lane & 15)) * ld + kk + ((lane >> 4) << 4);
  return *(const v16h*)p;                          // 16 contiguous halves
}

// ---------------- small utility kernels ----------------

__global__ void k_f2h(const float* __restrict__ s, half_t* __restrict__ d, int n) {
  int i = blockIdx.x * blockDim.x + threadIdx.x;
  if (i < n) d[i] = (half_t)s[i];
}

// stats: [sum(C) | sumsq(C)] -> scale/shift for y = x*scale + shift
__global__ void k_finalize(const float* __restrict__ stats, const float* __restrict__ g,
                           const float* __restrict__ b, float* __restrict__ scale,
                           float* __restrict__ shift, int C, float inv) {
  int c = blockIdx.x * blockDim.x + threadIdx.x;
  if (c >= C) return;
  float mean = stats[c] * inv;
  float var  = stats[C + c] * inv - mean * mean;
  float s = g[c] * rsqrtf(var + EPSV);
  scale[c] = s;
  shift[c] = b[c] - mean * s;
}

// ---------------- KNN: per-point top-21 by squared distance ----------------

__global__ __launch_bounds__(256) void k_knn(const float* __restrict__ xyz,
                                             int* __restrict__ idx) {
  __shared__ float sd[Npt];
  __shared__ float rv[256];
  __shared__ int   ri[256];
  int p = blockIdx.x;                // global point 0..8191
  int b = p >> 11;
  int t = threadIdx.x;
  float qx = xyz[(size_t)p * 3 + 0], qy = xyz[(size_t)p * 3 + 1], qz = xyz[(size_t)p * 3 + 2];
  const float* base = xyz + (size_t)b * Npt * 3;
  for (int m = t; m < Npt; m += 256) {
    float dx = base[m * 3 + 0] - qx, dy = base[m * 3 + 1] - qy, dz = base[m * 3 + 2] - qz;
    sd[m] = dx * dx + dy * dy + dz * dz;
  }
  __syncthreads();
  for (int kq = 0; kq < KNB; ++kq) {
    float bv = 3.3e38f; int bi = 0;
    for (int m = t; m < Npt; m += 256) {
      float v = sd[m];
      if (v < bv || (v == bv && m < bi)) { bv = v; bi = m; }
    }
    rv[t] = bv; ri[t] = bi; __syncthreads();
    for (int s = 128; s > 0; s >>= 1) {
      if (t < s) {
        if (rv[t + s] < rv[t] || (rv[t + s] == rv[t] && ri[t + s] < ri[t])) {
          rv[t] = rv[t + s]; ri[t] = ri[t + s];
        }
      }
      __syncthreads();
    }
    if (t == 0) { idx[(size_t)p * KNB + kq] = ri[0]; sd[ri[0]] = 3.4e38f; }
    __syncthreads();
  }
}

// ---------------- build feat_in (f16) + geo pre-activations + geo stats ---

__global__ __launch_bounds__(128) void k_build(
    const float* __restrict__ xyz, const float* __restrict__ f,
    const float* __restrict__ Wgeo, const int* __restrict__ idx,
    half_t* __restrict__ Afeat, float* __restrict__ geo_pre,
    float* __restrict__ stats_geo) {
  int p = blockIdx.x;            // point
  int c = threadIdx.x;           // 0..127 channel
  int b = p >> 11, n = p & (Npt - 1);
  float qx = xyz[(size_t)p * 3 + 0], qy = xyz[(size_t)p * 3 + 1], qz = xyz[(size_t)p * 3 + 2];
  float w0 = Wgeo[c * 6 + 0], w1 = Wgeo[c * 6 + 1], w2 = Wgeo[c * 6 + 2];
  float w3 = Wgeo[c * 6 + 3], w4 = Wgeo[c * 6 + 4], w5 = Wgeo[c * 6 + 5];
  float s = 0.f, s2 = 0.f;
  for (int k = 0; k < KNB; ++k) {
    size_t r = (size_t)p * KNB + k;
    int nb = idx[r];
    // feat_in row: [0:64] = neighbor features, [64:128] = center features
    float fv = (c < CinC) ? f[((size_t)b * CinC + c) * Npt + nb]
                          : f[((size_t)b * CinC + (c - CinC)) * Npt + n];
    Afeat[r * 128 + c] = (half_t)fv;
    size_t nbp = ((size_t)b * Npt + nb) * 3;
    float dx = xyz[nbp + 0] - qx, dy = xyz[nbp + 1] - qy, dz = xyz[nbp + 2] - qz;
    float g = w0 * qx + w1 * qy + w2 * qz + w3 * dx + w4 * dy + w5 * dz;
    geo_pre[r * 128 + c] = g;
    s += g; s2 += g * g;
  }
  atomicAdd(&stats_geo[c], s);
  atomicAdd(&stats_geo[128 + c], s2);
}

// ---------------- generic WMMA GEMM: C[M,O] = A[M,K] * W[O,K]^T (+bias) ---
// block = 256 (8 waves); block tile = 128 rows x 128 cols; wave = 16 rows.
// W panel for the current 32-wide k-step is staged in LDS (shared by all waves)
// so B fragments come from ds_load_b128 and no W pointer registers survive.

__global__ __launch_bounds__(256) void k_gemm(
    const half_t* __restrict__ A, const half_t* __restrict__ W,
    float* __restrict__ C, int K, int O,
    const float* __restrict__ bias, float* __restrict__ stats) {
  __shared__ __attribute__((aligned(32))) half_t ldsB[128][48];  // 32 cols used, padded
  int w = threadIdx.x >> 5, lane = threadIdx.x & 31;
  size_t m0 = (size_t)blockIdx.x * 128 + (size_t)w * 16;
  int ob = blockIdx.y * 128;
  const half_t* Arow = A + m0 * K;
  v8f acc[8] = {};
  for (int kk = 0; kk < K; kk += 32) {
    __syncthreads();
    {   // stage W[ob..ob+127][kk..kk+31] -> ldsB (8 KB payload)
      int row = threadIdx.x >> 1;
      int seg = (threadIdx.x & 1) * 16;
      *(v16h*)&ldsB[row][seg] = *(const v16h*)(W + (size_t)(ob + row) * K + kk + seg);
    }
    __syncthreads();
    if (kk + 32 < K) __builtin_prefetch(Arow + (size_t)(kk + 32), 0, 3);
    v16h va = load_a_frag(Arow, K, kk);
#pragma unroll
    for (int t = 0; t < 8; ++t) {
      v16h vb = load_b_rowT(&ldsB[0][0], 48, 0, t * 16);
      acc[t] = wmma32(va, vb, acc[t]);
    }
  }
#pragma unroll
  for (int t = 0; t < 8; ++t) {
    int o = ob + t * 16 + (lane & 15);
    float bs = bias ? bias[o] : 0.f;
    float s = 0.f, s2 = 0.f;
#pragma unroll
    for (int i = 0; i < 8; ++i) {
      size_t m = m0 + ((lane < 16) ? i : i + 8);
      float v = acc[t][i] + bs;
      C[m * O + o] = v;
      s += v; s2 += v * v;
    }
    if (stats) { atomicAdd(&stats[o], s); atomicAdd(&stats[O + o], s2); }
  }
}

// ---------------- BN+ReLU over K, maxpool -> feature_, stats for next BN --

__global__ void k_pool(const float* __restrict__ geo_pre, const float* __restrict__ fea_pre,
                       const float* __restrict__ scg, const float* __restrict__ scf,
                       float* __restrict__ feature, float* __restrict__ stats_bn) {
  int p = blockIdx.x, c = threadIdx.x;   // c 0..255
  const float* src; int ch; float scale, shift;
  if (c < 128) { src = geo_pre; ch = c;       scale = scg[ch]; shift = scg[128 + ch]; }
  else         { src = fea_pre; ch = c - 128; scale = scf[ch]; shift = scf[128 + ch]; }
  float m = 0.f;  // relu(...) >= 0, so max over K >= 0
  size_t base = (size_t)p * KNB * 128 + ch;
  for (int k = 0; k < KNB; ++k) {
    float v = fmaxf(src[base + (size_t)k * 128] * scale + shift, 0.f);
    m = fmaxf(m, v);
  }
  feature[(size_t)p * CoutC + c] = m;
  atomicAdd(&stats_bn[c], m);
  atomicAdd(&stats_bn[CoutC + c], m * m);
}

// x = bn(feature_) -> f16 (no relu)
__global__ void k_xh(const float* __restrict__ feature, const float* __restrict__ scb,
                     half_t* __restrict__ xh) {
  int p = blockIdx.x, c = threadIdx.x;
  size_t o = (size_t)p * CoutC + c;
  xh[o] = (half_t)(feature[o] * scb[c] + scb[CoutC + c]);
}

// relu(bn(pre)) -> f16  (q/k/v activations)
__global__ void k_bnrelu_h(const float* __restrict__ pre, const float* __restrict__ sc,
                           half_t* __restrict__ out) {
  int p = blockIdx.x, c = threadIdx.x;
  size_t o = (size_t)p * CoutC + c;
  out[o] = (half_t)fmaxf(pre[o] * sc[c] + sc[CoutC + c], 0.f);
}

// ---------------- flash attention, 7 shifted heads, WMMA everywhere -------
// grid = (N/64, HEADS, B); block = 128 (4 waves x 16 q-rows)

__global__ __launch_bounds__(128) void k_attn(
    const half_t* __restrict__ q16, const half_t* __restrict__ k16,
    const half_t* __restrict__ v16, half_t* __restrict__ Oh) {
  __shared__ __attribute__((aligned(32))) half_t ldsK[32][80];      // [key][ch], padded
  __shared__ __attribute__((aligned(32))) half_t ldsVt[64][48];     // [ch][key], padded
  __shared__ __attribute__((aligned(32))) half_t ldsP[4][16][40];   // per-wave P tile
  int h = blockIdx.y;
  int b = blockIdx.z;
  int w = threadIdx.x >> 5;
  int lane = threadIdx.x & 31;
  int chq = h * 32;                                 // shifted input slice base
  size_t rowbase = (size_t)b * Npt;
  int q0 = blockIdx.x * 64 + w * 16;

  const half_t* Qp = q16 + (rowbase + q0) * CoutC + chq;
  v16h aq0 = load_a_frag(Qp, CoutC, 0);
  v16h aq1 = load_a_frag(Qp, CoutC, 32);

  float mrun[8], lrun[8];
  v8f acc[4] = {};
#pragma unroll
  for (int i = 0; i < 8; ++i) { mrun[i] = -3.0e38f; lrun[i] = 0.f; }

  for (int kt = 0; kt < Npt; kt += 32) {
    __syncthreads();
    {   // stage K tile (vectorized) and V tile transposed (scalar)
      int key = threadIdx.x >> 2;
      int c0  = (threadIdx.x & 3) * 16;
      *(v16h*)&ldsK[key][c0] =
          *(const v16h*)(k16 + (rowbase + kt + key) * CoutC + chq + c0);
#pragma unroll 4
      for (int i = 0; i < 16; ++i) {
        int e = threadIdx.x * 16 + i; int kk = e >> 6; int cc = e & 63;
        ldsVt[cc][kk] = v16[(rowbase + kt + kk) * CoutC + chq + cc];
      }
    }
    __syncthreads();

    // S = Q * K^T over the 32-key tile (two 16-key column tiles)
    v8f s0 = {}, s1 = {};
    s0 = wmma32(aq0, load_b_rowT(&ldsK[0][0], 80, 0, 0), s0);
    s0 = wmma32(aq1, load_b_rowT(&ldsK[0][0], 80, 32, 0), s0);
    s1 = wmma32(aq0, load_b_rowT(&ldsK[0][0], 80, 0, 16), s1);
    s1 = wmma32(aq1, load_b_rowT(&ldsK[0][0], 80, 32, 16), s1);

    float alpha[8];
#pragma unroll
    for (int i = 0; i < 8; ++i) {
      float v = fmaxf(s0[i], s1[i]);
      v = fmaxf(v, __shfl_xor(v, 1)); v = fmaxf(v, __shfl_xor(v, 2));
      v = fmaxf(v, __shfl_xor(v, 4)); v = fmaxf(v, __shfl_xor(v, 8));
      float mn = fmaxf(mrun[i], v);
      alpha[i] = __expf(mrun[i] - mn);
      float p0 = __expf(s0[i] - mn), p1 = __expf(s1[i] - mn);
      float rs = p0 + p1;
      rs += __shfl_xor(rs, 1); rs += __shfl_xor(rs, 2);
      rs += __shfl_xor(rs, 4); rs += __shfl_xor(rs, 8);
      lrun[i] = lrun[i] * alpha[i] + rs;
      mrun[i] = mn;
      int row = (lane < 16) ? i : i + 8;                 // C-frag row for this lane
      ldsP[w][row][lane & 15] = (half_t)p0;
      ldsP[w][row][(lane & 15) + 16] = (half_t)p1;
    }
#pragma unroll
    for (int t = 0; t < 4; ++t)
#pragma unroll
      for (int i = 0; i < 8; ++i) acc[t][i] *= alpha[i];

    asm volatile("s_wait_dscnt 0" ::: "memory");         // P stores -> A-frag loads
    v16h ap = load_a_frag(&ldsP[w][0][0], 40, 0);
#pragma unroll
    for (int t = 0; t < 4; ++t)
      acc[t] = wmma32(ap, load_b_rowT(&ldsVt[0][0], 48, 0, t * 16), acc[t]);
  }

#pragma unroll
  for (int t = 0; t < 4; ++t) {
#pragma unroll
    for (int i = 0; i < 8; ++i) {
      int row = (lane < 16) ? i : i + 8;
      float val = acc[t][i] / lrun[i];
      Oh[(rowbase + q0 + row) * (HEADS * 64) + h * 64 + t * 16 + (lane & 15)] =
          (half_t)val;
    }
  }
}

// ---------------- fusion BN+ReLU, residual, LayerNorm -> y (f16) ----------

__global__ void k_fusln(const float* __restrict__ fus_pre, const float* __restrict__ scfus,
                        const float* __restrict__ feature, const float* __restrict__ g_ln,
                        const float* __restrict__ b_ln, float* __restrict__ fpp,
                        half_t* __restrict__ yh) {
  __shared__ float red[256];
  int p = blockIdx.x, c = threadIdx.x;
  size_t o = (size_t)p * CoutC + c;
  float v = fmaxf(fus_pre[o] * scfus[c] + scfus[CoutC + c], 0.f) + feature[o];
  fpp[o] = v;
  red[c] = v; __syncthreads();
  for (int s = 128; s > 0; s >>= 1) { if (c < s) red[c] += red[c + s]; __syncthreads(); }
  float mean = red[0] * (1.f / 256.f);
  __syncthreads();
  float d = v - mean;
  red[c] = d * d; __syncthreads();
  for (int s = 128; s > 0; s >>= 1) { if (c < s) red[c] += red[c + s]; __syncthreads(); }
  float var = red[0] * (1.f / 256.f);
  yh[o] = (half_t)(d * rsqrtf(var + EPSV) * g_ln[c] + b_ln[c]);
}

// ---------------- final residual + transpose to [B,Cout,N] ----------------

__global__ void k_out(const float* __restrict__ fpp, const float* __restrict__ mlp_pre,
                      float* __restrict__ out) {
  int p = blockIdx.x, c = threadIdx.x;
  int b = p >> 11, n = p & (Npt - 1);
  size_t o = (size_t)p * CoutC + c;
  out[((size_t)b * CoutC + c) * Npt + n] = fpp[o] + mlp_pre[o];
}

// ---------------- host orchestration ----------------

extern "C" void kernel_launch(void* const* d_in, const int* in_sizes, int n_in,
                              void* d_out, int out_size, void* d_ws, size_t ws_size,
                              hipStream_t stream) {
  const float* xyz      = (const float*)d_in[0];
  const float* f        = (const float*)d_in[1];
  const float* W_geo    = (const float*)d_in[2];
  const float* g_geo    = (const float*)d_in[3];
  const float* b_geo    = (const float*)d_in[4];
  const float* W_feat   = (const float*)d_in[5];
  const float* g_feat   = (const float*)d_in[6];
  const float* b_feat   = (const float*)d_in[7];
  const float* g_bn     = (const float*)d_in[8];
  const float* b_bn     = (const float*)d_in[9];
  const float* W_q      = (const float*)d_in[10];
  const float* g_q      = (const float*)d_in[11];
  const float* b_q      = (const float*)d_in[12];
  const float* W_k      = (const float*)d_in[13];
  const float* g_k      = (const float*)d_in[14];
  const float* b_k      = (const float*)d_in[15];
  const float* W_v      = (const float*)d_in[16];
  const float* g_v      = (const float*)d_in[17];
  const float* b_v      = (const float*)d_in[18];
  const float* W_fus    = (const float*)d_in[19];
  const float* bias_fus = (const float*)d_in[20];
  const float* g_fus    = (const float*)d_in[21];
  const float* b_fus    = (const float*)d_in[22];
  const float* g_ln     = (const float*)d_in[23];
  const float* b_ln     = (const float*)d_in[24];
  const float* W_mlp    = (const float*)d_in[25];
  const float* bias_mlp = (const float*)d_in[26];
  float* out = (float*)d_out;

  char* ws = (char*)d_ws;
  size_t off = 0;
  auto alloc = [&](size_t bytes) -> void* {
    void* p = ws + off; off = (off + bytes + 255) & ~(size_t)255; return p;
  };
  int*    idx      = (int*)   alloc((size_t)ROWS * sizeof(int));
  half_t* Afeat    = (half_t*)alloc((size_t)ROWS * 128 * sizeof(half_t));
  float*  geo_pre  = (float*) alloc((size_t)ROWS * 128 * sizeof(float));
  float*  fea_pre  = (float*) alloc((size_t)ROWS * 128 * sizeof(float));
  float*  feature  = (float*) alloc((size_t)NPT * CoutC * sizeof(float));
  half_t* xh       = (half_t*)alloc((size_t)NPT * CoutC * sizeof(half_t));
  float*  q_pre    = (float*) alloc((size_t)NPT * CoutC * sizeof(float));
  float*  k_pre    = (float*) alloc((size_t)NPT * CoutC * sizeof(float));
  float*  v_pre    = (float*) alloc((size_t)NPT * CoutC * sizeof(float));
  half_t* q16      = (half_t*)alloc((size_t)NPT * CoutC * sizeof(half_t));
  half_t* k16      = (half_t*)alloc((size_t)NPT * CoutC * sizeof(half_t));
  half_t* v16      = (half_t*)alloc((size_t)NPT * CoutC * sizeof(half_t));
  half_t* Oh       = (half_t*)alloc((size_t)NPT * HEADS * 64 * sizeof(half_t));
  float*  fus_pre  = (float*) alloc((size_t)NPT * CoutC * sizeof(float));
  float*  fpp      = (float*) alloc((size_t)NPT * CoutC * sizeof(float));
  half_t* yh       = (half_t*)alloc((size_t)NPT * CoutC * sizeof(half_t));
  float*  mlp_pre  = (float*) alloc((size_t)NPT * CoutC * sizeof(float));
  half_t* Wfeat_h  = (half_t*)alloc(128 * 128 * sizeof(half_t));
  half_t* Wq_h     = (half_t*)alloc(256 * 256 * sizeof(half_t));
  half_t* Wk_h     = (half_t*)alloc(256 * 256 * sizeof(half_t));
  half_t* Wv_h     = (half_t*)alloc(256 * 256 * sizeof(half_t));
  half_t* Wfus_h   = (half_t*)alloc(256 * 448 * sizeof(half_t));
  half_t* Wmlp_h   = (half_t*)alloc(256 * 256 * sizeof(half_t));
  float*  stats    = (float*) alloc(3072 * sizeof(float));
  float*  sc       = (float*) alloc(3072 * sizeof(float));

  // stats sub-buffers: [sum(C) | sumsq(C)] each
  float *sg = stats, *sf = stats + 256, *sb = stats + 512, *sq = stats + 1024,
        *sk = stats + 1536, *sv = stats + 2048, *sfus = stats + 2560;
  // scale/shift sub-buffers: [scale(C) | shift(C)] each
  float *scg = sc, *scf = sc + 256, *scb = sc + 512, *scq = sc + 1024,
        *sck = sc + 1536, *scv = sc + 2048, *scfus = sc + 2560;

  hipMemsetAsync(stats, 0, 3072 * sizeof(float), stream);

  // weight f32 -> f16
  k_f2h<<<(128 * 128 + 255) / 256, 256, 0, stream>>>(W_feat, Wfeat_h, 128 * 128);
  k_f2h<<<(256 * 256 + 255) / 256, 256, 0, stream>>>(W_q, Wq_h, 256 * 256);
  k_f2h<<<(256 * 256 + 255) / 256, 256, 0, stream>>>(W_k, Wk_h, 256 * 256);
  k_f2h<<<(256 * 256 + 255) / 256, 256, 0, stream>>>(W_v, Wv_h, 256 * 256);
  k_f2h<<<(256 * 448 + 255) / 256, 256, 0, stream>>>(W_fus, Wfus_h, 256 * 448);
  k_f2h<<<(256 * 256 + 255) / 256, 256, 0, stream>>>(W_mlp, Wmlp_h, 256 * 256);

  // KNN + gather/build + geo pre-activation
  k_knn<<<NPT, 256, 0, stream>>>(xyz, idx);
  k_build<<<NPT, 128, 0, stream>>>(xyz, f, W_geo, idx, Afeat, geo_pre, sg);

  // feat GEMM: [172032,128] x [128,128]^T  (+ stats)
  k_gemm<<<dim3((int)(ROWS / 128), 1), 256, 0, stream>>>(Afeat, Wfeat_h, fea_pre,
                                                         128, 128, nullptr, sf);
  const float inv_rows = 1.f / (float)ROWS;
  const float inv_pts  = 1.f / (float)NPT;
  k_finalize<<<1, 128, 0, stream>>>(sg, g_geo, b_geo, scg, scg + 128, 128, inv_rows);
  k_finalize<<<1, 128, 0, stream>>>(sf, g_feat, b_feat, scf, scf + 128, 128, inv_rows);

  // BN+ReLU + maxpool over K -> feature_, stats for next BN
  k_pool<<<NPT, 256, 0, stream>>>(geo_pre, fea_pre, scg, scf, feature, sb);
  k_finalize<<<1, 256, 0, stream>>>(sb, g_bn, b_bn, scb, scb + 256, 256, inv_pts);
  k_xh<<<NPT, 256, 0, stream>>>(feature, scb, xh);

  // Q/K/V projections (WMMA) + stats
  k_gemm<<<dim3(NPT / 128, 2), 256, 0, stream>>>(xh, Wq_h, q_pre, 256, 256, nullptr, sq);
  k_gemm<<<dim3(NPT / 128, 2), 256, 0, stream>>>(xh, Wk_h, k_pre, 256, 256, nullptr, sk);
  k_gemm<<<dim3(NPT / 128, 2), 256, 0, stream>>>(xh, Wv_h, v_pre, 256, 256, nullptr, sv);
  k_finalize<<<1, 256, 0, stream>>>(sq, g_q, b_q, scq, scq + 256, 256, inv_pts);
  k_finalize<<<1, 256, 0, stream>>>(sk, g_k, b_k, sck, sck + 256, 256, inv_pts);
  k_finalize<<<1, 256, 0, stream>>>(sv, g_v, b_v, scv, scv + 256, 256, inv_pts);
  k_bnrelu_h<<<NPT, 256, 0, stream>>>(q_pre, scq, q16);
  k_bnrelu_h<<<NPT, 256, 0, stream>>>(k_pre, sck, k16);
  k_bnrelu_h<<<NPT, 256, 0, stream>>>(v_pre, scv, v16);

  // flash attention over N=2048 per (batch, head)
  k_attn<<<dim3(Npt / 64, HEADS, Bsz), 128, 0, stream>>>(q16, k16, v16, Oh);

  // fusion GEMM [8192,448] x [256,448]^T + bias (+ stats)
  k_gemm<<<dim3(NPT / 128, 2), 256, 0, stream>>>(Oh, Wfus_h, fus_pre, 448, 256,
                                                 bias_fus, sfus);
  k_finalize<<<1, 256, 0, stream>>>(sfus, g_fus, b_fus, scfus, scfus + 256, 256, inv_pts);

  // BN+ReLU, residual, LayerNorm -> y (f16)
  k_fusln<<<NPT, 256, 0, stream>>>(fus_pre, scfus, feature, g_ln, b_ln, fpp, yh);

  // MLP GEMM + bias (no stats)
  k_gemm<<<dim3(NPT / 128, 2), 256, 0, stream>>>(yh, Wmlp_h, mlp_pre, 256, 256,
                                                 bias_mlp, nullptr);

  // final residual + transpose to [B, Cout, N]
  k_out<<<NPT, 256, 0, stream>>>(fpp, mlp_pre, out);
}
